// SimpleTrixFFN_20547123544584
// MI455X (gfx1250) — matrix operand order repaired and checked by reading, not compiled, not run on hardware
//
#include <hip/hip_runtime.h>

#define B_ 4096
#define D_ 1024
#define H_ 4096
#define T_ 8
#define C_ 1000
#define CP_ 1024   // classifier N padded to multiple of 256
#define LDT_ 40    // padded LDS row stride in bf16 elements (80B = 20 banks, conflict-free)

typedef __attribute__((ext_vector_type(16))) __bf16 v16bf;
typedef __attribute__((ext_vector_type(8)))  float  v8f;
typedef int v4i_t __attribute__((ext_vector_type(4)));
typedef __attribute__((address_space(1))) v4i_t as1_v4i;   // global
typedef __attribute__((address_space(3))) v4i_t as3_v4i;   // LDS

struct U4x2 { uint4 a, b; };

#if __has_builtin(__builtin_amdgcn_global_load_async_to_lds_b128) && \
    __has_builtin(__builtin_amdgcn_s_wait_asynccnt)
#define TRIX_ASYNC 1
#else
#define TRIX_ASYNC 0
#endif

__device__ __forceinline__ void trix_cp16(const void* g, void* l) {
#if TRIX_ASYNC
    __builtin_amdgcn_global_load_async_to_lds_b128((as1_v4i*)g, (as3_v4i*)l, 0, 0);
#else
    *reinterpret_cast<uint4*>(l) = *reinterpret_cast<const uint4*>(g);
#endif
}
__device__ __forceinline__ void trix_wait_async_prev() {   // current buf done, next in flight
#if TRIX_ASYNC
    __builtin_amdgcn_s_wait_asynccnt(6);
#endif
}
__device__ __forceinline__ void trix_wait_async_all() {
#if TRIX_ASYNC
    __builtin_amdgcn_s_wait_asynccnt(0);
#endif
}

__device__ __forceinline__ unsigned short f2bf(float f) {
    unsigned u = __float_as_uint(f);
    u = (u + 0x7FFFu + ((u >> 16) & 1u)) >> 16;   // round-to-nearest-even
    return (unsigned short)u;
}

// ---------------- routing: scores, softmax(T=0.5), argmax -------------------
__global__ __launch_bounds__(256) void trix_route(const float* __restrict__ x,
                                                  const float* __restrict__ sig,
                                                  float* __restrict__ wgt_ws,
                                                  float* __restrict__ w_out,
                                                  int* __restrict__ idx_out) {
    const int wave = blockIdx.x * 8 + (threadIdx.x >> 5);
    const int lane = threadIdx.x & 31;
    if (wave >= B_) return;
    const float* xr = x + (size_t)wave * D_;
    float s[T_] = {};
    for (int d = lane; d < D_; d += 32) {
        const float xv = xr[d];
#pragma unroll
        for (int t = 0; t < T_; t++) s[t] += xv * sig[t * D_ + d];
    }
#pragma unroll
    for (int off = 16; off > 0; off >>= 1)
#pragma unroll
        for (int t = 0; t < T_; t++) s[t] += __shfl_xor(s[t], off, 32);
    if (lane == 0) {
        float mx = s[0]; int mi = 0;
#pragma unroll
        for (int t = 1; t < T_; t++) if (s[t] > mx) { mx = s[t]; mi = t; }
        float e[T_], sum = 0.f;
#pragma unroll
        for (int t = 0; t < T_; t++) { e[t] = expf(2.0f * (s[t] - mx)); sum += e[t]; }
        const float inv = 1.0f / sum;
#pragma unroll
        for (int t = 0; t < T_; t++) {
            const float wv = e[t] * inv;
            wgt_ws[wave * T_ + t] = wv;
            w_out[wave * T_ + t]  = wv;
        }
        idx_out[wave] = mi;
    }
}

// ---------------- conversions ----------------------------------------------
__global__ __launch_bounds__(256) void trix_cvt_bf(const float* __restrict__ src,
                                                   unsigned short* __restrict__ dst, int n) {
    const int i = blockIdx.x * 256 + threadIdx.x;
    if (i < n) dst[i] = f2bf(src[i]);
}

__global__ __launch_bounds__(256) void trix_cvt_x_init(const float* __restrict__ x,
                                                       unsigned short* __restrict__ xb,
                                                       float* __restrict__ acc, int n) {
    const int i = blockIdx.x * 256 + threadIdx.x;
    if (i < n) { const float v = x[i]; xb[i] = f2bf(v); acc[i] = v; }  // residual init
}

__global__ __launch_bounds__(256) void trix_cvt_wcls(const float* __restrict__ src,
                                                     unsigned short* __restrict__ dst, int n) {
    const int i = blockIdx.x * 256 + threadIdx.x;
    if (i < n) {
        const int row = i >> 10;                      // D_ = 1024
        dst[i] = (row < C_) ? f2bf(src[i]) : (unsigned short)0;
    }
}

// ---------------- WMMA GEMM: C[M,N] = A[M,K] @ W[N,K]^T --------------------
// LDS-staged via async global->LDS copies, double buffered.
// MODE 0: h = gelu(C + bias)           -> bf16 out (ld = N)
// MODE 1: acc += rowScale[m]*(C+bias)  -> f32 accumulate (ld = N)
// MODE 2: logits = C + bias            -> f32 out, cols < outN
template <int MODE>
__global__ __launch_bounds__(256) void trix_gemm(
    const unsigned short* __restrict__ A, const unsigned short* __restrict__ W,
    const float* __restrict__ bias, const float* __restrict__ rowScale, int t,
    unsigned short* __restrict__ outBf, float* __restrict__ outAcc,
    float* __restrict__ outF, int K, int N, int outN) {

    __shared__ unsigned short lA[2][128 * LDT_];   // 2 x 10240 B
    __shared__ unsigned short lB[2][256 * LDT_];   // 2 x 20480 B   (total 60 KB)

    const int tid  = threadIdx.x;
    const int lane = tid & 31;
    const int wave = tid >> 5;      // 8 waves: 2 (M) x 4 (N)
    const int half = lane >> 4;
    const int ln15 = lane & 15;
    const int wM = wave >> 2, wN = wave & 3;
    const int m0b = blockIdx.x * 128;
    const int n0b = blockIdx.y * 256;

    const unsigned short* Ag = A + (size_t)m0b * K;
    const unsigned short* Bg = W + (size_t)n0b * K;

    // stage one 32-wide K slab: A 128x32 (512 x 16B chunks), B 256x32 (1024 chunks)
    auto copy_tile = [&](int buf, int k0) {
#pragma unroll
        for (int i = 0; i < 2; i++) {
            const int c = tid + 256 * i, row = c >> 2, sub = c & 3;
            trix_cp16(Ag + (size_t)row * K + k0 + sub * 8, &lA[buf][row * LDT_ + sub * 8]);
        }
#pragma unroll
        for (int i = 0; i < 4; i++) {
            const int c = tid + 256 * i, row = c >> 2, sub = c & 3;
            trix_cp16(Bg + (size_t)row * K + k0 + sub * 8, &lB[buf][row * LDT_ + sub * 8]);
        }
    };

    v8f acc[4][4] = {};
    const int nIter = K >> 5;
    copy_tile(0, 0);

    for (int it = 0; it < nIter; ++it) {
        const int cur = it & 1;
        if (it + 1 < nIter) {
            copy_tile(cur ^ 1, (it + 1) << 5);   // prefetch next slab (6 async ops/wave)
            trix_wait_async_prev();              // <=6 outstanding => current slab landed
        } else {
            trix_wait_async_all();
        }
        __syncthreads();

        v16bf a[4], b[4];
#pragma unroll
        for (int i = 0; i < 4; i++) {            // A frag: 2 x 16B at +0 / +32B
            const unsigned short* p = &lA[cur][(wM * 64 + i * 16 + ln15) * LDT_ + half * 8];
            U4x2 q;
            q.a = *reinterpret_cast<const uint4*>(p);
            q.b = *reinterpret_cast<const uint4*>(p + 16);
            a[i] = __builtin_bit_cast(v16bf, q);
        }
#pragma unroll
        for (int j = 0; j < 4; j++) {            // B frag: 2 x 16B at +0 / +32B
            const unsigned short* p = &lB[cur][(wN * 64 + j * 16 + ln15) * LDT_ + half * 8];
            U4x2 q;
            q.a = *reinterpret_cast<const uint4*>(p);
            q.b = *reinterpret_cast<const uint4*>(p + 16);
            b[j] = __builtin_bit_cast(v16bf, q);
        }
#pragma unroll
        for (int i = 0; i < 4; i++)
#pragma unroll
            for (int j = 0; j < 4; j++)
                acc[i][j] = __builtin_amdgcn_wmma_f32_16x16x32_bf16(
                    false, a[i], false, b[j], (short)0, acc[i][j], false, false);

        __syncthreads();                         // protect cur^1 before next overwrite
    }

    const int m0 = m0b + wM * 64;
    const int n0 = n0b + wN * 64;
#pragma unroll
    for (int j = 0; j < 4; j++) {
        const int n = n0 + j * 16 + ln15;
        const float bj = (MODE == 2) ? ((n < outN) ? bias[n] : 0.0f) : bias[n];
#pragma unroll
        for (int i = 0; i < 4; i++) {
#pragma unroll
            for (int e = 0; e < 8; e++) {
                const int m = m0 + i * 16 + half * 8 + e;
                const float c = acc[i][j][e] + bj;
                if (MODE == 0) {
                    const float g = 0.5f * c * (1.0f + erff(c * 0.70710678118654752f));
                    outBf[(size_t)m * N + n] = f2bf(g);
                } else if (MODE == 1) {
                    outAcc[(size_t)m * N + n] += c * rowScale[m * T_ + t];
                } else {
                    if (n < outN) outF[(size_t)m * outN + n] = c;
                }
            }
        }
    }
}

// ---------------- launch ----------------------------------------------------
extern "C" void kernel_launch(void* const* d_in, const int* in_sizes, int n_in,
                              void* d_out, int out_size, void* d_ws, size_t ws_size,
                              hipStream_t stream) {
    const float* x      = (const float*)d_in[0];
    const float* sig    = (const float*)d_in[1];
    const float* W_up   = (const float*)d_in[2];
    const float* b_up   = (const float*)d_in[3];
    const float* W_down = (const float*)d_in[4];
    const float* b_down = (const float*)d_in[5];
    const float* W_cls  = (const float*)d_in[6];
    const float* b_cls  = (const float*)d_in[7];

    float* logits = (float*)d_out;                                 // (B, C) f32
    int*   idxOut = (int*)((char*)d_out + (size_t)B_ * C_ * 4);    // (B,)   i32
    float* wOut   = (float*)(idxOut + B_);                         // (B, T) f32

    char* w = (char*)d_ws;
    unsigned short* xb   = (unsigned short*)w; w += (size_t)B_ * D_ * 2;   // 8 MB
    unsigned short* hb   = (unsigned short*)w; w += (size_t)B_ * H_ * 2;   // 32 MB
    unsigned short* wub  = (unsigned short*)w; w += (size_t)H_ * D_ * 2;   // 8 MB
    unsigned short* wdb  = (unsigned short*)w; w += (size_t)D_ * H_ * 2;   // 8 MB
    unsigned short* wcb  = (unsigned short*)w; w += (size_t)CP_ * D_ * 2;  // 2 MB
    unsigned short* outb = (unsigned short*)w; w += (size_t)B_ * D_ * 2;   // 8 MB
    float* accBuf        = (float*)w;          w += (size_t)B_ * D_ * 4;   // 16 MB
    float* wgt           = (float*)w;          w += (size_t)B_ * T_ * 4;   // 128 KB

    trix_route<<<B_ / 8, 256, 0, stream>>>(x, sig, wgt, wOut, idxOut);
    trix_cvt_x_init<<<(B_ * D_ + 255) / 256, 256, 0, stream>>>(x, xb, accBuf, B_ * D_);
    trix_cvt_wcls<<<(CP_ * D_ + 255) / 256, 256, 0, stream>>>(W_cls, wcb, CP_ * D_);

    for (int t = 0; t < T_; t++) {
        trix_cvt_bf<<<(H_ * D_ + 255) / 256, 256, 0, stream>>>(
            W_up + (size_t)t * H_ * D_, wub, H_ * D_);
        trix_gemm<0><<<dim3(B_ / 128, H_ / 256), 256, 0, stream>>>(
            xb, wub, b_up + t * H_, nullptr, t, hb, nullptr, nullptr, D_, H_, H_);
        trix_cvt_bf<<<(D_ * H_ + 255) / 256, 256, 0, stream>>>(
            W_down + (size_t)t * D_ * H_, wdb, D_ * H_);
        trix_gemm<1><<<dim3(B_ / 128, D_ / 256), 256, 0, stream>>>(
            hb, wdb, b_down + t * D_, wgt, t, nullptr, accBuf, nullptr, H_, D_, D_);
    }

    trix_cvt_bf<<<(B_ * D_ + 255) / 256, 256, 0, stream>>>(accBuf, outb, B_ * D_);
    trix_gemm<2><<<dim3(B_ / 128, CP_ / 256), 256, 0, stream>>>(
        outb, wcb, b_cls, nullptr, 0, nullptr, nullptr, logits, D_, CP_, C_);
}
